// GraphCastEdgeBlock_21801253994713
// MI455X (gfx1250) — matrix operand order repaired and checked by reading, not compiled, not run on hardware
//
#include <hip/hip_runtime.h>
#include <math.h>

typedef _Float16 v16h __attribute__((ext_vector_type(16)));
typedef _Float16 v8h  __attribute__((ext_vector_type(8)));
typedef _Float16 v4h  __attribute__((ext_vector_type(4)));
typedef float    v8f  __attribute__((ext_vector_type(8)));

#define LN_EPS 1e-5f
#define MT     128    // edges per block
#define MTILES 8      // 16-row WMMA tiles per block

// Build a 16-element f16 fragment from two contiguous 8-element (16B) chunks.
static __device__ __forceinline__ v16h frag16(const _Float16* p0, const _Float16* p1) {
  v8h lo = *(const v8h*)p0;
  v8h hi = *(const v8h*)p1;
  v16h r;
#pragma unroll
  for (int i = 0; i < 8; ++i) { r[i] = lo[i]; r[i + 8] = hi[i]; }
  return r;
}

// ---- Prep: W1 [384,512] f32 -> Wt1 [512][384] f16 (N-major);
//            W2 [512,128] f32 -> Wt2 [128][512] f16 (N-major). ----
__global__ void prep_weights(const float* __restrict__ W1, const float* __restrict__ W2,
                             unsigned short* __restrict__ Wt1u,
                             unsigned short* __restrict__ Wt2u) {
  int idx = blockIdx.x * blockDim.x + threadIdx.x;
  _Float16* Wt1 = (_Float16*)Wt1u;
  _Float16* Wt2 = (_Float16*)Wt2u;
  if (idx < 512 * 384) {
    int n = idx / 384, k = idx % 384;
    Wt1[idx] = (_Float16)W1[k * 512 + n];
  } else if (idx < 512 * 384 + 128 * 512) {
    int i = idx - 512 * 384;
    int n = i / 512, k = i % 512;
    Wt2[i] = (_Float16)W2[k * 128 + n];
  }
}

// LDS: region0 [0, 133120): A1 f16 [128][392] ALIASED with A2 f16 [128][520]
//      region1 [133120, 135168): sum[128], sumsq[128], mu[128], rs[128]
__global__ __launch_bounds__(1024)
void edge_block_kernel(const float* __restrict__ edge_attr,
                       const float* __restrict__ nsrc,
                       const float* __restrict__ ndst,
                       const long long* __restrict__ eidx,
                       const float* __restrict__ b1,
                       const float* __restrict__ gam,
                       const float* __restrict__ bet,
                       const float* __restrict__ b2,
                       const unsigned short* __restrict__ Wt1u,
                       const unsigned short* __restrict__ Wt2u,
                       float* __restrict__ out, int E) {
  const _Float16* Wt1 = (const _Float16*)Wt1u;
  const _Float16* Wt2 = (const _Float16*)Wt2u;

  __shared__ __align__(16) unsigned char smem[133120 + 2048];
  _Float16* A1 = (_Float16*)smem;             // [128][392] f16
  _Float16* A2 = (_Float16*)smem;             // [128][520] f16 (aliased; A1 dead)
  float* sumS  = (float*)(smem + 133120);     // [128]
  float* sum2S = sumS + 128;                  // [128]
  float* muS   = sum2S + 128;                 // [128]
  float* rsS   = muS + 128;                   // [128]

  const int tid  = threadIdx.x;
  const int lane = tid & 31;
  const int wave = tid >> 5;     // 0..31
  const int m    = lane & 15;    // A row within tile / B,C column within tile
  const int half = lane >> 4;
  const int tileBase = blockIdx.x * MT;

  if (tid < 256) sumS[tid] = 0.0f;   // zero sum + sumsq (256 contiguous floats)

  // ---- Phase 0: gather + concat + f32->f16 into A tile [128 x 384] ----
  {
    int r   = tid >> 3;              // 0..127 : edge row
    int sub = tid & 7;               // 8 threads per row, 12 float4 each
    int e   = tileBase + r;
    int eg  = e < E ? e : (E - 1);
    long long si = eidx[eg];
    long long di = eidx[(long long)E + eg];
    const float* rowA = edge_attr + (long long)eg * 128;
    const float* rowS = nsrc + si * 128;
    const float* rowD = ndst + di * 128;
#pragma unroll
    for (int q = 0; q < 12; ++q) {
      int f = sub * 12 + q;          // float4 index 0..95
      int c = f * 4;
      float4 v;
      if (c < 128)      v = *(const float4*)(rowA + c);
      else if (c < 256) v = *(const float4*)(rowS + (c - 128));
      else              v = *(const float4*)(rowD + (c - 256));
      v4h b;
      b[0] = (_Float16)v.x; b[1] = (_Float16)v.y;
      b[2] = (_Float16)v.z; b[3] = (_Float16)v.w;
      *(v4h*)(A1 + r * 392 + c) = b;
    }
  }
  __syncthreads();

  // ---- Phase 1: GEMM1  h[128,512] = A[128,384] x W1 ----
  // Wave owns ONE 16-column N-tile; B fragment reused across 8 M-tiles.
  // B for next K-step is prefetched before the WMMA burst; A fragments are
  // preloaded in groups of 4 so ds latency overlaps the matrix pipe.
  const int col0 = wave * 16 + m;
  v8f acc[MTILES] = {};
  {
    const _Float16* bbase = Wt1 + col0 * 384 + half * 8;
    v16h bcur = frag16(bbase, bbase + 16);
    for (int ks = 0; ks < 12; ++ks) {
      v16h bnext = bcur;
      if (ks < 11) {
        const _Float16* bp = bbase + (ks + 1) * 32;
        bnext = frag16(bp, bp + 16);
      }
#pragma unroll
      for (int g = 0; g < 2; ++g) {
        v16h af[4];
#pragma unroll
        for (int i = 0; i < 4; ++i) {
          const _Float16* ap = A1 + ((g * 4 + i) * 16 + m) * 392 + ks * 32 + half * 8;
          af[i] = frag16(ap, ap + 16);
        }
#pragma unroll
        for (int i = 0; i < 4; ++i) {
          acc[g * 4 + i] = __builtin_amdgcn_wmma_f32_16x16x32_f16(
              false, af[i], false, bcur, (short)0, acc[g * 4 + i], false, false);
        }
      }
      bcur = bnext;
    }
  }

  // bias add (per column)
  {
    float bb0 = b1[col0];
#pragma unroll
    for (int mt = 0; mt < MTILES; ++mt)
#pragma unroll
      for (int j = 0; j < 8; ++j) acc[mt][j] += bb0;
  }

  // ---- Phase 2: LayerNorm stats straight from C-layout registers ----
  // Lane holds rows mt*16 + half*8 + j, column col0. Reduce across the 16 lanes
  // of each half-wave via shfl_xor, then ds_add_f32 per row.
#pragma unroll
  for (int mt = 0; mt < MTILES; ++mt) {
#pragma unroll
    for (int j = 0; j < 8; ++j) {
      float v  = acc[mt][j];
      float v2 = v * v;
#pragma unroll
      for (int o = 1; o < 16; o <<= 1) {
        v  += __shfl_xor(v, o, 32);
        v2 += __shfl_xor(v2, o, 32);
      }
      if (m == 0) {
        int row = mt * 16 + half * 8 + j;
        atomicAdd(&sumS[row], v);
        atomicAdd(&sum2S[row], v2);
      }
    }
  }
  __syncthreads();   // stats complete; also: everyone past GEMM1 -> A1 dead

  if (tid < MT) {
    float mu  = sumS[tid] * (1.0f / 512.0f);
    float var = sum2S[tid] * (1.0f / 512.0f) - mu * mu;
    muS[tid] = mu;
    rsS[tid] = rsqrtf(var + LN_EPS);
  }
  __syncthreads();

  // ---- Phase 3: LN + SiLU from registers -> A2 f16 [128][520] (over A1) ----
  {
    float ga0 = gam[col0], be0 = bet[col0];
#pragma unroll
    for (int mt = 0; mt < MTILES; ++mt) {
#pragma unroll
      for (int j = 0; j < 8; ++j) {
        int row = mt * 16 + half * 8 + j;
        float mu = muS[row], rs = rsS[row];
        float y = (acc[mt][j] - mu) * rs * ga0 + be0;
        float s = y / (1.0f + __expf(-y));
        A2[row * 520 + col0] = (_Float16)s;
      }
    }
  }
  __syncthreads();

  // ---- Phase 4: GEMM2  out[128,128] = A2[128,512] x W2 ----
  // 8 N-tiles x 8 M-tiles = 64 tasks over waves 0..15 (1 N-tile x 4 M-tiles each);
  // B fragment reused by 4 M-tiles, Wt2 read only twice per block.
  if (wave < 16) {
    const int nt2  = wave & 7;        // N-tile 0..7
    const int mtg  = wave >> 3;       // M-tile group 0..1 -> tiles mtg*4 .. mtg*4+3
    const int col2 = nt2 * 16 + m;
    v8f acc2[4] = {};
    {
      const _Float16* bbase = Wt2 + col2 * 512 + half * 8;
      v16h bcur = frag16(bbase, bbase + 16);
      for (int ks = 0; ks < 16; ++ks) {
        v16h bnext = bcur;
        if (ks < 15) {
          const _Float16* bp = bbase + (ks + 1) * 32;
          bnext = frag16(bp, bp + 16);
        }
        v16h af[4];
#pragma unroll
        for (int i = 0; i < 4; ++i) {
          const _Float16* ap =
              A2 + ((mtg * 4 + i) * 16 + m) * 520 + ks * 32 + half * 8;
          af[i] = frag16(ap, ap + 16);
        }
#pragma unroll
        for (int i = 0; i < 4; ++i) {
          acc2[i] = __builtin_amdgcn_wmma_f32_16x16x32_f16(
              false, af[i], false, bcur, (short)0, acc2[i], false, false);
        }
        bcur = bnext;
      }
    }

    // ---- Phase 5: bias + residual + store ----
    float bb2 = b2[col2];
#pragma unroll
    for (int i = 0; i < 4; ++i) {
#pragma unroll
      for (int j = 0; j < 8; ++j) {
        int row = (mtg * 4 + i) * 16 + half * 8 + j;
        int e = tileBase + row;
        if (e < E) {
          long long off = (long long)e * 128 + col2;
          out[off] = acc2[i][j] + bb2 + edge_attr[off];
        }
      }
    }
  }
}

extern "C" void kernel_launch(void* const* d_in, const int* in_sizes, int n_in,
                              void* d_out, int out_size, void* d_ws, size_t ws_size,
                              hipStream_t stream) {
  const float*     edge_attr = (const float*)d_in[0];
  const float*     nsrc      = (const float*)d_in[1];
  const float*     ndst      = (const float*)d_in[2];
  const long long* eidx      = (const long long*)d_in[3];   // int64 [2,E]
  const float*     W1        = (const float*)d_in[4];
  const float*     b1        = (const float*)d_in[5];
  const float*     gam       = (const float*)d_in[6];
  const float*     bet       = (const float*)d_in[7];
  const float*     W2        = (const float*)d_in[8];
  const float*     b2        = (const float*)d_in[9];
  float* out = (float*)d_out;

  int E = in_sizes[0] / 128;

  unsigned short* Wt1 = (unsigned short*)d_ws;          // 512*384 f16 = 384 KB
  unsigned short* Wt2 = Wt1 + 512 * 384;                // 128*512 f16 = 128 KB

  int prepTotal = 512 * 384 + 128 * 512;
  prep_weights<<<(prepTotal + 255) / 256, 256, 0, stream>>>(W1, W2, Wt1, Wt2);

  int tiles = (E + MT - 1) / MT;
  edge_block_kernel<<<tiles, 1024, 0, stream>>>(edge_attr, nsrc, ndst, eidx,
                                                b1, gam, bet, b2, Wt1, Wt2, out, E);
}
